// AgentCommunicationModule_18700287607227
// MI455X (gfx1250) — compile-verified, matrix operand order checked by reading, not run on hardware
//
#include <hip/hip_runtime.h>
#include <hip/hip_bf16.h>

#define NN 4096
#define IN_F 128
#define OUT_F 128
#define NH 4
#define HD 32
#define SLOPE 0.2f
#define LN_EPS 1e-5f
#define LOG2E 1.44269504f

typedef __bf16 bf16_t;
typedef __attribute__((ext_vector_type(16))) __bf16 v16bf;
typedef __attribute__((ext_vector_type(8)))  __bf16 v8bf;
typedef __attribute__((ext_vector_type(8)))  float  v8f;

// ---------------- Kernel 1: Wx = x @ W^T (f32 exact) + bf16 transposed copy ----
__global__ void k_wx(const float* __restrict__ x, const float* __restrict__ W,
                     float* __restrict__ Wx, bf16_t* __restrict__ WxT) {
  const int n = blockIdx.x;
  const int f = threadIdx.x;
  __shared__ float xs[IN_F];
  xs[f] = x[(size_t)n * IN_F + f];
  __syncthreads();
  const float* wr = W + (size_t)f * IN_F;
  float acc = 0.f;
#pragma unroll 8
  for (int k = 0; k < IN_F; ++k) acc = fmaf(xs[k], wr[k], acc);
  Wx[(size_t)n * OUT_F + f] = acc;
  WxT[(size_t)f * NN + n] = (bf16_t)acc;   // d-major for WMMA B-operand gathering
}

// ---------------- Kernel 2: per-node attention logit halves -------------------
__global__ void k_alpha(const float* __restrict__ Wx, const float* __restrict__ a,
                        float* __restrict__ ai, float* __restrict__ aj) {
  const int n = blockIdx.x * blockDim.x + threadIdx.x;
  if (n >= NN) return;
  const float* wr = Wx + (size_t)n * OUT_F;
#pragma unroll
  for (int h = 0; h < NH; ++h) {
    float si = 0.f, sj = 0.f;
    const float* ah = a + h * 2 * HD;
#pragma unroll 8
    for (int d = 0; d < HD; ++d) {
      const float w = wr[h * HD + d];
      si = fmaf(w, ah[d], si);
      sj = fmaf(w, ah[HD + d], sj);
    }
    ai[h * NN + n] = si;
    aj[h * NN + n] = sj;
  }
}

// ---------------- Kernel 3: per-head max of alpha_j (softmax stabilizer) ------
__global__ void k_ajmax(const float* __restrict__ aj, float* __restrict__ ajmax) {
  const int h = blockIdx.x;
  __shared__ float red[256];
  float m = -1e30f;
  for (int n = threadIdx.x; n < NN; n += 256) m = fmaxf(m, aj[h * NN + n]);
  red[threadIdx.x] = m;
  __syncthreads();
  for (int s = 128; s > 0; s >>= 1) {
    if (threadIdx.x < (unsigned)s)
      red[threadIdx.x] = fmaxf(red[threadIdx.x], red[threadIdx.x + s]);
    __syncthreads();
  }
  if (threadIdx.x == 0) ajmax[h] = red[0];
}

// ---------------- Kernel 4: fused masked-softmax attention aggregate ----------
// One block = 16 output rows; 8 waves = 4 heads x 2 j-range halves.
// Per j-tile (32 cols) each wave issues THREE v_wmma_f32_16x16x32_bf16:
//   acc0/acc1 : P(16x32) @ Wx(32x16) for d=0..15 / d=16..31
//   accz      : P @ ones  -> per-row softmax denominator on the XDL pipe
// A-operand K-swizzle (16-bit A 16x32): lane L: M=L&15,
//   element e -> K = (e>=8?16:0) + 8*(L>>4) + (e&7)
// Loop trip count is a compile-time constant so loop control is scalar and
// EXEC stays all-ones through every WMMA (ISA requirement).
__global__ void __launch_bounds__(256)
k_gat(const unsigned char* __restrict__ adj,
      const float* __restrict__ x,
      const float* __restrict__ ai_all,
      const float* __restrict__ aj_all,
      const float* __restrict__ ajmax,
      const bf16_t* __restrict__ WxT,
      const float* __restrict__ gamma,
      const float* __restrict__ beta,
      float* __restrict__ out) {
  const int i0   = blockIdx.x * 16;
  const int tid  = threadIdx.x;
  const int w    = tid >> 5;
  const int h    = w & 3;          // head
  const int jh   = w >> 2;         // j-range half (0 or 1)
  const int lane = tid & 31;
  const int m    = lane & 15;      // WMMA M row
  const int hi   = lane >> 4;      // K-half selector
  const int half8 = hi << 3;

  __shared__ float lds_acc[NH][24][32];   // partner partials: acc0|acc1|accz
  __shared__ float lds_out[16][OUT_F];
  __shared__ float lds_s[16][16];
  __shared__ float lds_s2[16][16];
  __shared__ float lds_mu[16];
  __shared__ float lds_ri[16];

  const int irow = i0 + m;
  const float* ajh = aj_all + h * NN;
  const float aim = ai_all[h * NN + irow];
  const float t0 = aim + ajmax[h];
  const float MrowC = fmaxf(t0, SLOPE * t0) * LOG2E;  // stabilizer, log2 domain
  // exp2( lrelu(aim+aj)*LOG2E - MrowC ) = exp2( max(fma(aj,C,c1), fma(aj,sC,c2)) )
  const float c1 = aim * LOG2E - MrowC;
  const float c2 = SLOPE * aim * LOG2E - MrowC;

  const unsigned char* adjRow = adj + (size_t)irow * NN;
  const bf16_t* wrow0 = WxT + (size_t)(h * HD + m) * NN;        // d = m
  const bf16_t* wrow1 = wrow0 + (size_t)16 * NN;                // d = m+16

  v16bf Bones;
#pragma unroll
  for (int e = 0; e < 16; ++e) Bones[e] = (bf16_t)1.0f;

  v8f acc0 = {};   // d 0..15
  v8f acc1 = {};   // d 16..31
  v8f accz = {};   // row sums (softmax denominator)

  const int jbeg = jh * (NN / 2);
  for (int it = 0; it < (NN / 2) / 32; ++it) {      // constant trip count
    const int j0 = jbeg + it * 32;
    __builtin_prefetch(adjRow + j0 + 1024, 0, 3);   // global_prefetch_b8
    // adjacency: 0/1 bytes for this lane's 16 K slots (two 8-byte runs)
    const uint2 c0 = *(const uint2*)(adjRow + j0 + half8);
    const uint2 c1w = *(const uint2*)(adjRow + j0 + 16 + half8);
    // alpha_j for the same 16 K slots
    const float4 q0 = *(const float4*)(ajh + j0 + half8);
    const float4 q1 = *(const float4*)(ajh + j0 + half8 + 4);
    const float4 q2 = *(const float4*)(ajh + j0 + 16 + half8);
    const float4 q3 = *(const float4*)(ajh + j0 + 16 + half8 + 4);
    const float av[16] = {q0.x, q0.y, q0.z, q0.w, q1.x, q1.y, q1.z, q1.w,
                          q2.x, q2.y, q2.z, q2.w, q3.x, q3.y, q3.z, q3.w};
    const unsigned mw[4] = {c0.x, c0.y, c1w.x, c1w.y};
    v16bf A;
#pragma unroll
    for (int e = 0; e < 16; ++e) {
      const float u = fmaf(av[e], LOG2E, c1);
      const float v = fmaf(av[e], SLOPE * LOG2E, c2);
      float p = __builtin_amdgcn_exp2f(fmaxf(u, v));
      // byte is 0/1 -> multiply masks; v_cvt_f32_ubyteN extracts+converts
      p *= (float)((mw[e >> 2] >> ((e & 3) * 8)) & 0xffu);
      A[e] = (bf16_t)p;
    }
    // B operands (bf16, K-swizzled contiguous 16B chunks from d-major WxT)
    const v8bf b0lo = *(const v8bf*)(wrow0 + j0 + half8);
    const v8bf b0hi = *(const v8bf*)(wrow0 + j0 + 16 + half8);
    const v8bf b1lo = *(const v8bf*)(wrow1 + j0 + half8);
    const v8bf b1hi = *(const v8bf*)(wrow1 + j0 + 16 + half8);
    const v16bf B0 = __builtin_shufflevector(b0lo, b0hi, 0, 1, 2, 3, 4, 5, 6, 7,
                                             8, 9, 10, 11, 12, 13, 14, 15);
    const v16bf B1 = __builtin_shufflevector(b1lo, b1hi, 0, 1, 2, 3, 4, 5, 6, 7,
                                             8, 9, 10, 11, 12, 13, 14, 15);
    acc0 = __builtin_amdgcn_wmma_f32_16x16x32_bf16(false, A, false, B0,
                                                   (short)0, acc0, false, false);
    acc1 = __builtin_amdgcn_wmma_f32_16x16x32_bf16(false, A, false, B1,
                                                   (short)0, acc1, false, false);
    accz = __builtin_amdgcn_wmma_f32_16x16x32_bf16(false, A, false, Bones,
                                                   (short)0, accz, false, false);
  }

  // ---- combine the two j-halves: jh==1 publishes, jh==0 accumulates ----
  if (jh == 1) {
#pragma unroll
    for (int r = 0; r < 8; ++r) {
      lds_acc[h][r][lane]      = acc0[r];
      lds_acc[h][8 + r][lane]  = acc1[r];
      lds_acc[h][16 + r][lane] = accz[r];
    }
  }
  __syncthreads();
  if (jh == 0) {
    // scale by 1/Z and stage the 16x(4*32) tile.
    // C layout: VGPR r -> row = r + 8*(lane>>4), col = lane&15; accz[r] holds
    // the matching row's denominator in the SAME lane.
#pragma unroll
    for (int r = 0; r < 8; ++r) {
      const int row = r + (hi << 3);
      const float zr = accz[r] + lds_acc[h][16 + r][lane];
      const float rz = __builtin_amdgcn_rcpf(zr);
      lds_out[row][h * HD + m]      = (acc0[r] + lds_acc[h][r][lane]) * rz;
      lds_out[row][h * HD + 16 + m] = (acc1[r] + lds_acc[h][8 + r][lane]) * rz;
    }
  }
  __syncthreads();

  // ---- residual + LayerNorm: 16 threads per row, 8 features each ----
  const int row = tid >> 4, sub = tid & 15;
  const float* xrow = x + (size_t)(i0 + row) * IN_F;
  float s = 0.f, s2 = 0.f;
#pragma unroll
  for (int q = 0; q < 8; ++q) {
    const int f = sub * 8 + q;
    const float y = lds_out[row][f] + xrow[f];
    lds_out[row][f] = y;
    s += y;
    s2 += y * y;
  }
  lds_s[row][sub] = s;
  lds_s2[row][sub] = s2;
  __syncthreads();
  if (tid < 16) {
    float ts = 0.f, ts2 = 0.f;
#pragma unroll
    for (int u = 0; u < 16; ++u) { ts += lds_s[tid][u]; ts2 += lds_s2[tid][u]; }
    const float mu = ts / OUT_F;
    lds_mu[tid] = mu;
    lds_ri[tid] = rsqrtf(ts2 / OUT_F - mu * mu + LN_EPS);
  }
  __syncthreads();
  const float mu = lds_mu[row], ri = lds_ri[row];
  float* orow = out + (size_t)(i0 + row) * OUT_F;
#pragma unroll
  for (int q = 0; q < 8; ++q) {
    const int f = sub * 8 + q;
    orow[f] = gamma[f] * (lds_out[row][f] - mu) * ri + beta[f];
  }
}

// ---------------- launcher ----------------------------------------------------
extern "C" void kernel_launch(void* const* d_in, const int* in_sizes, int n_in,
                              void* d_out, int out_size, void* d_ws, size_t ws_size,
                              hipStream_t stream) {
  (void)in_sizes; (void)n_in; (void)out_size; (void)ws_size;
  const float*         x     = (const float*)d_in[0];
  const unsigned char* adj   = (const unsigned char*)d_in[1];   // jnp.bool_ -> 1 byte
  const float*         W     = (const float*)d_in[2];
  const float*         a     = (const float*)d_in[3];
  const float*         gamma = (const float*)d_in[4];
  const float*         beta  = (const float*)d_in[5];
  float*               out   = (float*)d_out;

  char* ws = (char*)d_ws;
  float*  Wx    = (float*)(ws);                          // 4096*128*4   = 2 MiB
  bf16_t* WxT   = (bf16_t*)(ws + 2097152);               // 128*4096*2   = 1 MiB
  float*  ai    = (float*)(ws + 3145728);                // 4*4096*4     = 64 KiB
  float*  aj    = (float*)(ws + 3211264);                // 4*4096*4     = 64 KiB
  float*  ajmax = (float*)(ws + 3276800);                // 16 B

  k_wx<<<NN, IN_F, 0, stream>>>(x, W, Wx, WxT);
  k_alpha<<<NN / 128, 128, 0, stream>>>(Wx, a, ai, aj);
  k_ajmax<<<NH, 256, 0, stream>>>(aj, ajmax);
  k_gat<<<NN / 16, 256, 0, stream>>>(adj, x, ai, aj, ajmax, WxT, gamma, beta, out);
}
